// Encode_64579128262771
// MI455X (gfx1250) — compile-verified, matrix operand order checked by reading, not run on hardware
//
#include <hip/hip_runtime.h>
#include <hip/hip_bf16.h>

// ---------------------------------------------------------------------------
// Stacked LSTM (Keras-default) for MI455X / gfx1250, wave32 + WMMA bf16.
//   B=128, T=256, H=512, 6 passes total (1 + 4 shared-weight + 1 final).
// Strategy: one fused kernel per timestep. Each wave computes a 16x16
// (batch x unit) block for ALL FOUR gates (4 f32 accumulators sharing the
// A fragment), then applies sigmoid/tanh/c/h updates fully in-register.
// Weights pre-packed into the CDNA5 WMMA B-fragment layout (32B/lane loads).
// h is double-buffered (read t-1, write t) to avoid intra-kernel races.
// ---------------------------------------------------------------------------

typedef __attribute__((ext_vector_type(16))) __bf16 v16bf;
typedef __attribute__((ext_vector_type(8)))  __bf16 v8bf;
typedef __attribute__((ext_vector_type(8)))  float  v8f;

#define BATCH 128
#define SEQL  256
#define HID   512
#define G4    2048   // 4*HID

union FragU { v16bf v; v8bf h[2]; };

__device__ __forceinline__ float sigm(float x) { return 1.0f / (1.0f + __expf(-x)); }

// ---------------------------------------------------------------------------
// Pack f32 weight [K=512, N=2048] (row-major) into bf16 WMMA B-fragments.
// Fragment for (n-tile nt, k-chunk kc): lane l holds 16 contiguous bf16 where
// element j corresponds to  k = kc*32 + (j/8)*16 + (l/16)*8 + (j%8),
//                           n = nt*16 + (l%16)
// (mirrors the documented 16-bit matrix lane/VGPR striping, §7.12.2).
// ---------------------------------------------------------------------------
__global__ __launch_bounds__(256) void pack_weight_kernel(
    const float* __restrict__ W, __bf16* __restrict__ P)
{
    int idx = blockIdx.x * 256 + threadIdx.x;          // 0 .. 2^20-1
    int j  =  idx        & 15;
    int l  = (idx >> 4)  & 31;
    int kc = (idx >> 9)  & 15;
    int nt =  idx >> 13;                               // 0..127
    int k  = kc * 32 + ((j >> 3) << 4) + ((l >> 4) << 3) + (j & 7);
    int n  = nt * 16 + (l & 15);
    P[idx] = (__bf16)W[(size_t)k * G4 + n];
}

// f32 -> bf16 bulk convert (input sequence x)
__global__ __launch_bounds__(256) void convert_bf16_kernel(
    const float* __restrict__ src, __bf16* __restrict__ dst, int n)
{
    for (int i = blockIdx.x * 256 + threadIdx.x; i < n; i += gridDim.x * 256)
        dst[i] = (__bf16)src[i];
}

// zero initial h (bf16) and c (f32): 128*512 = 65536 elements each
__global__ __launch_bounds__(256) void init_state_kernel(
    __bf16* __restrict__ h, float* __restrict__ c)
{
    int i = blockIdx.x * 256 + threadIdx.x;
    if (i < BATCH * HID) { h[i] = (__bf16)0.0f; c[i] = 0.0f; }
}

// ---------------------------------------------------------------------------
// One LSTM timestep, fully fused:  z = x_t@W + h_{t-1}@U + b ; gates ; c,h.
// Grid: 32 blocks x 256 threads (8 waves).  256 waves total.
// Wave w: mtile = w/32 (16 batch rows), utile = w%32 (16 hidden units).
// acc[g] accumulates gate g (n = g*512 + u) -> i,f,g,o land in matching
// accumulator slots for the same (b,u), so the nonlinearity is elementwise.
// ---------------------------------------------------------------------------
__global__ __launch_bounds__(256) void lstm_step_kernel(
    const __bf16* __restrict__ Xin,   // [B, T, H] bf16 (pass input sequence)
    int t,
    const __bf16* __restrict__ Wp,    // packed [128 nt][16 kc][32 lane][16] bf16
    const __bf16* __restrict__ Up,    // packed recurrent weights
    const float*  __restrict__ bias,  // [2048] f32
    const __bf16* __restrict__ Hprev, // [B, H] bf16  (h_{t-1}, read-only)
    __bf16*       __restrict__ Hnext, // [B, H] bf16  (h_t, write)
    float*        __restrict__ Cbuf,  // [B, H] f32   (c, read-modify-write)
    __bf16*       __restrict__ SeqOut,// [B, T, H] bf16 or nullptr
    float*        __restrict__ OutF)  // [B, H] f32 final output or nullptr
{
    const int lane  = threadIdx.x & 31;
    const int wave  = threadIdx.x >> 5;
    const int w     = blockIdx.x * 8 + wave;   // 0..255
    const int mtile = w >> 5;                  // 0..7
    const int utile = w & 31;                  // 0..31
    const int m0    = mtile * 16;
    const int u0    = utile * 16;
    const int half  = lane >> 4;               // 0 or 1
    const int arow  = m0 + (lane & 15);        // batch row this lane loads (A)

    v8f acc[4] = {};                           // i, f, g, o accumulators

    // ---- phase 1: x_t @ W  (K = 512, 16 chunks of 32) ----
    const __bf16* xrow = Xin + ((size_t)arow * SEQL + t) * HID;
    #pragma unroll 2
    for (int kc = 0; kc < 16; ++kc) {
        FragU a;
        a.h[0] = *reinterpret_cast<const v8bf*>(xrow + kc * 32 + half * 8);
        a.h[1] = *reinterpret_cast<const v8bf*>(xrow + kc * 32 + 16 + half * 8);
        // warm L2/L0 for the recurrent-weight fragments used in phase 2
        __builtin_prefetch(Up + (((size_t)utile * 16 + kc) * 32 + lane) * 16, 0, 1);
        #pragma unroll
        for (int g = 0; g < 4; ++g) {
            const int nt = g * 32 + utile;     // gate g -> n = g*512 + u
            const v16bf bfr = *reinterpret_cast<const v16bf*>(
                Wp + (((size_t)nt * 16 + kc) * 32 + lane) * 16);
            acc[g] = __builtin_amdgcn_wmma_f32_16x16x32_bf16(
                false, a.v, false, bfr, (short)0, acc[g], false, false);
        }
    }

    // ---- phase 2: h_{t-1} @ U  (K = 512) ----
    const __bf16* hrow = Hprev + (size_t)arow * HID;
    #pragma unroll 2
    for (int kc = 0; kc < 16; ++kc) {
        FragU a;
        a.h[0] = *reinterpret_cast<const v8bf*>(hrow + kc * 32 + half * 8);
        a.h[1] = *reinterpret_cast<const v8bf*>(hrow + kc * 32 + 16 + half * 8);
        #pragma unroll
        for (int g = 0; g < 4; ++g) {
            const int nt = g * 32 + utile;
            const v16bf bfr = *reinterpret_cast<const v16bf*>(
                Up + (((size_t)nt * 16 + kc) * 32 + kc * 0 + lane) * 16);
            acc[g] = __builtin_amdgcn_wmma_f32_16x16x32_bf16(
                false, a.v, false, bfr, (short)0, acc[g], false, false);
        }
    }

    // ---- epilogue: gates + state update, elementwise in-register ----
    // C/D layout: VGPR r -> M = r (lanes 0-15) / r+8 (lanes 16-31); N = lane%16
    const int u_abs = u0 + (lane & 15);
    const float bi = bias[u_abs];
    const float bf = bias[u_abs + 512];
    const float bg = bias[u_abs + 1024];
    const float bo = bias[u_abs + 1536];

    #pragma unroll
    for (int r = 0; r < 8; ++r) {
        const int brow = m0 + half * 8 + r;
        const size_t off = (size_t)brow * HID + u_abs;
        const float zi = acc[0][r] + bi;
        const float zf = acc[1][r] + bf;
        const float zg = acc[2][r] + bg;
        const float zo = acc[3][r] + bo;
        const float ig = sigm(zi);
        const float fg = sigm(zf);
        const float gg = tanhf(zg);
        const float og = sigm(zo);
        const float cn = fg * Cbuf[off] + ig * gg;
        const float hv = og * tanhf(cn);
        Cbuf[off]  = cn;
        Hnext[off] = (__bf16)hv;
        if (SeqOut) SeqOut[((size_t)brow * SEQL + t) * HID + u_abs] = (__bf16)hv;
        if (OutF)   OutF[off] = hv;
    }
}

// ---------------------------------------------------------------------------
extern "C" void kernel_launch(void* const* d_in, const int* in_sizes, int n_in,
                              void* d_out, int out_size, void* d_ws, size_t ws_size,
                              hipStream_t stream)
{
    (void)in_sizes; (void)n_in; (void)out_size; (void)ws_size;
    const float* x  = (const float*)d_in[0];
    const float* W1 = (const float*)d_in[1];
    const float* U1 = (const float*)d_in[2];
    const float* b1 = (const float*)d_in[3];
    const float* W2 = (const float*)d_in[4];
    const float* U2 = (const float*)d_in[5];
    const float* b2 = (const float*)d_in[6];

    char* ws = (char*)d_ws;
    size_t off = 0;
    auto carve = [&](size_t bytes) -> void* {
        void* p = ws + off;
        off += (bytes + 255) & ~(size_t)255;
        return p;
    };

    const size_t seq_elems = (size_t)BATCH * SEQL * HID;   // 16,777,216
    const size_t wgt_elems = (size_t)HID * G4;             // 1,048,576
    __bf16* xbf  = (__bf16*)carve(seq_elems * 2);
    __bf16* bufA = (__bf16*)carve(seq_elems * 2);
    __bf16* bufB = (__bf16*)carve(seq_elems * 2);
    __bf16* wp1  = (__bf16*)carve(wgt_elems * 2);
    __bf16* up1  = (__bf16*)carve(wgt_elems * 2);
    __bf16* wp2  = (__bf16*)carve(wgt_elems * 2);
    __bf16* up2  = (__bf16*)carve(wgt_elems * 2);
    __bf16* h0   = (__bf16*)carve((size_t)BATCH * HID * 2);
    __bf16* h1   = (__bf16*)carve((size_t)BATCH * HID * 2);
    float*  cb   = (float*) carve((size_t)BATCH * HID * 4);

    // one-time prep (re-done every call: deterministic, no cached state)
    pack_weight_kernel<<<4096, 256, 0, stream>>>(W1, wp1);
    pack_weight_kernel<<<4096, 256, 0, stream>>>(U1, up1);
    pack_weight_kernel<<<4096, 256, 0, stream>>>(W2, wp2);
    pack_weight_kernel<<<4096, 256, 0, stream>>>(U2, up2);
    convert_bf16_kernel<<<8192, 256, 0, stream>>>(x, xbf, (int)seq_elems);
    init_state_kernel<<<(BATCH * HID) / 256, 256, 0, stream>>>(h0, cb);

    // 6 passes: x->A, A->B, B->A, A->B, B->A, final reads A (h only)
    const __bf16* pin [6] = { xbf, bufA, bufB, bufA, bufB, bufA };
    __bf16*       pout[6] = { bufA, bufB, bufA, bufB, bufA, nullptr };

    __bf16* hp = h0;   // h_{t-1} (read)
    __bf16* hn = h1;   // h_t     (write)
    for (int p = 0; p < 6; ++p) {
        const __bf16* Wp = (p < 5) ? wp1 : wp2;
        const __bf16* Up = (p < 5) ? up1 : up2;
        const float*  bb = (p < 5) ? b1  : b2;
        for (int t = 0; t < SEQL; ++t) {
            float* of = (p == 5 && t == SEQL - 1) ? (float*)d_out : nullptr;
            lstm_step_kernel<<<32, 256, 0, stream>>>(
                pin[p], t, Wp, Up, bb, hp, hn, cb, pout[p], of);
            __bf16* tmp = hp; hp = hn; hn = tmp;   // double-buffer swap
        }
    }
}
